// GVP_6176162971780
// MI455X (gfx1250) — compile-verified, matrix operand order checked by reading, not run on hardware
//
#include <hip/hip_runtime.h>
#include <hip/hip_bf16.h>
#include <math.h>

// ---------------------------------------------------------------------------
// GVP layer for MI455X (gfx1250): bf16 WMMA, f32 accumulate.
// Each wave: 32 nodes (two 16-row tiles) sharing B fragments -> halves L2
// weight traffic, two independent WMMA accumulator chains for ILP.
// Streaming inputs/outputs use non-temporal hints; weights stay L2-resident.
// ---------------------------------------------------------------------------

typedef __bf16 bf16;
typedef __attribute__((ext_vector_type(16))) __bf16 v16bf;
typedef __attribute__((ext_vector_type(8)))  __bf16 v8bf;
typedef __attribute__((ext_vector_type(8)))  float  v8f;
typedef __attribute__((ext_vector_type(4)))  float  v4f;

#define N_NODES 262144
#define EPS 1e-4f

// d_ws layout (bf16 element offsets)
#define WS_WH   0                    // [32][32]
#define WS_WMU  1024                 // [32][32]
#define WS_WM   2048                 // [256][288]
#define WS_WG   (2048 + 73728)       // [32][256]
#define WS_TOTAL (WS_WG + 8192)      // 83968 bf16 = 167936 bytes

// per-wave LDS (bytes):
//   [0,16384)      : V staging (12288 used) then reused as sigS [2][16][256] bf16
//   [16384,22528)  : VhA [2][3][16][32] bf16
//   [22528,24576)  : shA [2][16][32] bf16
#define LDS_PER_WAVE 24576
#define WAVES_PER_BLOCK 4

__device__ __forceinline__ v8f wmma_bf16(v16bf a, v16bf b, v8f c) {
    return __builtin_amdgcn_wmma_f32_16x16x32_bf16(
        false, a, false, b, (short)0, c, false, false);
}

// Fragment from row-major bf16 [rows][stride]; A: row = node row,
// B: row = output column (weights are [out][K] row-major = B columns).
// Lane mapping: row = lane&15, K-halves at kb=(lane>>4)*8 and kb+16.
__device__ __forceinline__ v16bf frag_bf16(const bf16* base, int row, int stride, int koff) {
    const bf16* p = base + (size_t)row * stride + koff;
    v8bf lo = *(const v8bf*)(p);
    v8bf hi = *(const v8bf*)(p + 16);
    v16bf f;
#pragma unroll
    for (int i = 0; i < 8; ++i) { f[i] = lo[i]; f[i + 8] = hi[i]; }
    return f;
}

// A fragment from row-major f32 global memory (streaming: non-temporal loads).
__device__ __forceinline__ v16bf frag_f32_nt(const float* base, int row, int stride, int koff) {
    const float* p = base + (size_t)row * stride + koff;
    v4f a0 = __builtin_nontemporal_load((const v4f*)(p));
    v4f a1 = __builtin_nontemporal_load((const v4f*)(p + 4));
    v4f b0 = __builtin_nontemporal_load((const v4f*)(p + 16));
    v4f b1 = __builtin_nontemporal_load((const v4f*)(p + 20));
    v16bf f;
#pragma unroll
    for (int i = 0; i < 4; ++i) {
        f[i]      = (bf16)a0[i];
        f[i + 4]  = (bf16)a1[i];
        f[i + 8]  = (bf16)b0[i];
        f[i + 12] = (bf16)b1[i];
    }
    return f;
}

// ---- prep: fp32 weights -> bf16 in d_ws (once, stream-ordered) ----
__global__ void gvp_prep(const float* __restrict__ Wh, const float* __restrict__ Wmu,
                         const float* __restrict__ Wm, const float* __restrict__ Wg,
                         bf16* __restrict__ ws) {
    int i = blockIdx.x * blockDim.x + threadIdx.x;
    if (i < 1024)              ws[WS_WH  + i]           = (bf16)Wh[i];
    else if (i < 2048)         ws[WS_WMU + (i - 1024)]  = (bf16)Wmu[i - 1024];
    else if (i < 2048 + 73728) ws[WS_WM  + (i - 2048)]  = (bf16)Wm[i - 2048];
    else if (i < WS_TOTAL)     ws[WS_WG  + (i - 75776)] = (bf16)Wg[i - 75776];
}

// ---- main: one wave per 32-node (2x16) tile pair ----
__global__ void __launch_bounds__(WAVES_PER_BLOCK * 32)
gvp_main(const float* __restrict__ s, const float* __restrict__ V,
         const bf16* __restrict__ ws, const float* __restrict__ Wmb,
         const float* __restrict__ Wgb, float* __restrict__ out_s,
         float* __restrict__ out_v) {
    extern __shared__ char smem[];
    const int lane  = threadIdx.x & 31;
    const int wave  = threadIdx.x >> 5;
    const int node0 = (blockIdx.x * WAVES_PER_BLOCK + wave) * 32;

    char* wbase = smem + wave * LDS_PER_WAVE;
    float* Vt   = (float*)(wbase);             // [2][16][32][3] f32 (phase 1 only)
    bf16*  sigS = (bf16*)(wbase);              // [2][16][256] (phase 2+, reuses Vt)
    bf16*  VhA  = (bf16*)(wbase + 16384);      // [2][3][16][32] A-layout
    bf16*  shA  = (bf16*)(wbase + 22528);      // [2][16][32]    A-layout

    const int r  = lane & 15;          // A row / B column within tile
    const int kb = (lane >> 4) * 8;    // K half base
    const int mb = (lane >> 4) * 8;    // D row base (VGPR e -> row mb+e, col r)

    const bf16* wh  = ws + WS_WH;
    const bf16* wmu = ws + WS_WMU;
    const bf16* wm  = ws + WS_WM;
    const bf16* wg  = ws + WS_WG;

    // ---- stage V for 32 nodes (contiguous 12 KB) into LDS ----
    {
        const v4f* src = (const v4f*)(V + (size_t)node0 * 96);
        v4f* dst = (v4f*)Vt;
#pragma unroll
        for (int i = 0; i < 24; ++i)
            dst[i * 32 + lane] = __builtin_nontemporal_load(src + i * 32 + lane);
    }
    __syncthreads();

    // ---- per row tile u: V_h = V @ W_h^T ; norms -> shA ; V_h -> VhA ----
#pragma unroll
    for (int u = 0; u < 2; ++u) {
        v16bf aV[3];
#pragma unroll
        for (int c = 0; c < 3; ++c) {
            v16bf f;
#pragma unroll
            for (int j = 0; j < 8; ++j) {
                f[j]     = (bf16)Vt[u * 1536 + r * 96 + (kb + j) * 3 + c];
                f[j + 8] = (bf16)Vt[u * 1536 + r * 96 + (kb + 16 + j) * 3 + c];
            }
            aV[c] = f;
        }
#pragma unroll
        for (int t = 0; t < 2; ++t) {
            v8f vh[3];
#pragma unroll
            for (int c = 0; c < 3; ++c) {
                v16bf b = frag_bf16(wh, t * 16 + r, 32, kb);
                v8f z = {};
                vh[c] = wmma_bf16(aV[c], b, z);
            }
            const int h = t * 16 + r;
#pragma unroll
            for (int e = 0; e < 8; ++e) {
                const int m = mb + e;
                float x = vh[0][e], y = vh[1][e], z2 = vh[2][e];
                float nrm = fmaxf(sqrtf(x * x + y * y + z2 * z2), EPS);
                shA[u * 512 + m * 32 + h]             = (bf16)nrm;
                VhA[(u * 3 + 0) * 512 + m * 32 + h]   = (bf16)x;
                VhA[(u * 3 + 1) * 512 + m * 32 + h]   = (bf16)y;
                VhA[(u * 3 + 2) * 512 + m * 32 + h]   = (bf16)z2;
            }
        }
    }

    // ---- A fragments for K=288 concat input, both row tiles, in registers ----
    v16bf aS[2][9];
#pragma unroll
    for (int u = 0; u < 2; ++u)
#pragma unroll
        for (int k = 0; k < 8; ++k)
            aS[u][k] = frag_f32_nt(s + (size_t)(node0 + u * 16) * 256, r, 256, k * 32 + kb);
    __syncthreads();   // Vt region dead past here -> sigS may be written
#pragma unroll
    for (int u = 0; u < 2; ++u)
        aS[u][8] = frag_bf16(shA + u * 512, r, 32, kb);

    // ---- main GEMM: s_m[32x256], K=288; one B fragment feeds 2 acc chains ----
    for (int j = 0; j < 16; ++j) {
        v8f acc0 = {}, acc1 = {};
#pragma unroll
        for (int k = 0; k < 9; ++k) {
            v16bf b = frag_bf16(wm, j * 16 + r, 288, k * 32 + kb);
            acc0 = wmma_bf16(aS[0][k], b, acc0);
            acc1 = wmma_bf16(aS[1][k], b, acc1);
        }
        const int n = j * 16 + r;
        const float bias = Wmb[n];
#pragma unroll
        for (int u = 0; u < 2; ++u) {
            const v8f& acc = u ? acc1 : acc0;
#pragma unroll
            for (int e = 0; e < 8; ++e) {
                const int m = mb + e;
                float sm = acc[e] + bias;
                __builtin_nontemporal_store(
                    fmaxf(sm, 0.0f),
                    out_s + (size_t)(node0 + u * 16 + m) * 256 + n);
                float sg = 1.0f / (1.0f + __expf(-sm));
                sigS[u * 4096 + m * 256 + n] = (bf16)sg;
            }
        }
    }
    __syncthreads();

    // ---- gate GEMM (K=256) + V_mu GEMMs (K=32); D layouts coincide ----
    v16bf aG[2][8];
#pragma unroll
    for (int u = 0; u < 2; ++u)
#pragma unroll
        for (int k = 0; k < 8; ++k)
            aG[u][k] = frag_bf16(sigS + u * 4096, r, 256, k * 32 + kb);

#pragma unroll
    for (int t = 0; t < 2; ++t) {
        v8f g0 = {}, g1 = {};
#pragma unroll
        for (int k = 0; k < 8; ++k) {
            v16bf b = frag_bf16(wg, t * 16 + r, 256, k * 32 + kb);
            g0 = wmma_bf16(aG[0][k], b, g0);
            g1 = wmma_bf16(aG[1][k], b, g1);
        }
        v8f vm[2][3];
#pragma unroll
        for (int c = 0; c < 3; ++c) {
            v16bf b = frag_bf16(wmu, t * 16 + r, 32, kb);
#pragma unroll
            for (int u = 0; u < 2; ++u) {
                v16bf a = frag_bf16(VhA + (u * 3 + c) * 512, r, 32, kb);
                v8f z = {};
                vm[u][c] = wmma_bf16(a, b, z);
            }
        }
        const int n = t * 16 + r;
        const float gb = Wgb[n];
#pragma unroll
        for (int u = 0; u < 2; ++u) {
            const v8f& ga = u ? g1 : g0;
#pragma unroll
            for (int e = 0; e < 8; ++e) {
                const int m = mb + e;
                float g = 1.0f / (1.0f + __expf(-(ga[e] + gb)));
                float* o = out_v + (size_t)(node0 + u * 16 + m) * 96 + n * 3;
                __builtin_nontemporal_store(g * vm[u][0][e], o + 0);
                __builtin_nontemporal_store(g * vm[u][1][e], o + 1);
                __builtin_nontemporal_store(g * vm[u][2][e], o + 2);
            }
        }
    }
}

extern "C" void kernel_launch(void* const* d_in, const int* in_sizes, int n_in,
                              void* d_out, int out_size, void* d_ws, size_t ws_size,
                              hipStream_t stream) {
    const float* s   = (const float*)d_in[0];
    const float* V   = (const float*)d_in[1];
    const float* Wh  = (const float*)d_in[2];
    const float* Wmu = (const float*)d_in[3];
    const float* Wm  = (const float*)d_in[4];
    const float* Wmb = (const float*)d_in[5];
    const float* Wg  = (const float*)d_in[6];
    const float* Wgb = (const float*)d_in[7];
    float* out_s = (float*)d_out;
    float* out_v = out_s + (size_t)N_NODES * 256;
    bf16*  ws    = (bf16*)d_ws;   // needs 167936 bytes of scratch

    // 1) weights fp32 -> bf16 once (L2-resident, shared by all WGPs)
    gvp_prep<<<(WS_TOTAL + 255) / 256, 256, 0, stream>>>(Wh, Wmu, Wm, Wg, ws);
    // 2) main: 2048 blocks x 4 waves x 32 nodes = 262144 nodes exactly
    gvp_main<<<N_NODES / (WAVES_PER_BLOCK * 32), WAVES_PER_BLOCK * 32,
               WAVES_PER_BLOCK * LDS_PER_WAVE, stream>>>(
        s, V, ws, Wmb, Wgb, out_s, out_v);
}
